// HexCircleConv_14310831030367
// MI455X (gfx1250) — matrix operand order repaired and checked by reading, not compiled, not run on hardware
//
#include <hip/hip_runtime.h>

// ---------------------------------------------------------------------------
// HexCircleConv for MI455X (gfx1250, wave32, WMMA).
//
// out[b,o,n] = sum_{i,k} W[o,i,k] * x[b,i,nbr[n,k]]  (+ bias[o])
// = GEMM: M = B*N = 59360, 128 cols, reduction CIN*K = 2432, gathered A.
// 37 GFLOP; ~62 MB HBM; gather traffic served from 192 MB L2 (x resident).
//
// v3: on top of v2's 64-row / 4-accumulator register blocking (4x less
// packed-weight L2 traffic), software-pipeline the A-fragment LDS reads at
// distance 3: v2's codegen showed ds_load_b128 + s_wait_dscnt 0x0 directly
// before every WMMA (full LDS latency per matrix op).  Now each fragment's
// loads are issued 3 WMMA-steps ahead via 3 rotating register slots, and
// all 4 B fragments for a k-step are preloaded as one clause.
// ---------------------------------------------------------------------------

#define BB   32
#define CIN  128
#define COUT 128
#define NN   1855
#define KNB  19
#define MTOT (BB * NN)        // 59360
#define RT   4                // row-tiles (16 rows each) per workgroup

typedef __bf16 v16bf __attribute__((ext_vector_type(16)));
typedef float  v8f   __attribute__((ext_vector_type(8)));

union Frag { uint4 q[2]; v16bf v; };

__device__ __forceinline__ unsigned short f2bf(float f) {
  // round-to-nearest-even fp32 -> bf16
  unsigned u = __float_as_uint(f);
  u += 0x7FFFu + ((u >> 16) & 1u);
  return (unsigned short)(u >> 16);
}

// --------------------------------------------------------------------------
// Pre-kernel 1: x (B,CIN,N) f32 -> xT (B,N,CIN) bf16, LDS-tiled transpose.
// --------------------------------------------------------------------------
__global__ void transpose_x_kernel(const float* __restrict__ x,
                                   unsigned short* __restrict__ xT) {
  __shared__ float t[32][33];
  const int b  = blockIdx.z;
  const int i0 = blockIdx.y * 32;
  const int n0 = blockIdx.x * 32;
  const int tx = threadIdx.x, ty = threadIdx.y;   // (32, 8)
#pragma unroll
  for (int j = 0; j < 4; ++j) {
    const int i = i0 + ty + j * 8;
    const int n = n0 + tx;
    t[ty + j * 8][tx] = (n < NN) ? x[((size_t)b * CIN + i) * NN + n] : 0.f;
  }
  __syncthreads();
#pragma unroll
  for (int j = 0; j < 4; ++j) {
    const int n = n0 + ty + j * 8;
    const int i = i0 + tx;
    if (n < NN)
      xT[((size_t)b * NN + n) * CIN + i] = f2bf(t[tx][ty + j * 8]);
  }
}

// --------------------------------------------------------------------------
// Pre-kernel 2: pack weight (COUT,CIN,K) f32 into the WMMA B-fragment
// layout (bf16): B 32x16 (KxN), lanes 0-15 hold K=0..7,16..23, lanes 16-31
// hold K=8..15,24..31 (same K order as the A fragment).
// wPack idx = (((k*4 + ii)*8 + oT)*32 + lane)*16 + j
//   <- weight[o = oT*16 + lane%16][i = ii*32 + (lane>=16?8:0) + (j&7) + (j>=8?16:0)][k]
// --------------------------------------------------------------------------
__global__ void pack_w_kernel(const float* __restrict__ w,
                              unsigned short* __restrict__ wp) {
  const int idx = blockIdx.x * 256 + threadIdx.x;
  if (idx >= KNB * 4 * 8 * 32 * 16) return;
  const int j    = idx & 15;
  const int lane = (idx >> 4) & 31;
  const int oT   = (idx >> 9) & 7;
  const int ii   = (idx >> 12) & 3;
  const int k    = idx >> 14;
  const int o = oT * 16 + (lane & 15);
  const int i = ii * 32 + ((lane >= 16) ? 8 : 0) + (j & 7) + ((j >= 8) ? 16 : 0);
  wp[idx] = f2bf(w[((size_t)o * CIN + i) * KNB + k]);
}

// --------------------------------------------------------------------------
// Main kernel: one workgroup (8 wave32s) per 64-row M block (4 row-tiles).
// Wave w computes o-tile w for all 4 row-tiles, reusing each register-
// resident B fragment 4x.  A-fragment LDS reads pipelined at distance 3.
// --------------------------------------------------------------------------
#define APITCH 136   // ushort elements per LDS row (128 + 8 pad)
#define TSTRIDE (16 * APITCH)

__device__ __forceinline__ Frag ldA(const unsigned short* paBase, int s) {
  const int tt = s & 3, ii = s >> 2;
  const unsigned short* pa = paBase + tt * TSTRIDE + ii * 32;
  Frag f;
  f.q[0] = *(const uint4*)(pa);        // K = base + 0..7
  f.q[1] = *(const uint4*)(pa + 16);   // K = base + 16..23
  return f;
}

__global__ __launch_bounds__(256, 2)
void conv_main_kernel(const unsigned short* __restrict__ xT,
                      const unsigned short* __restrict__ wPack,
                      const int* __restrict__ nbr,
                      const float* __restrict__ bias,
                      float* __restrict__ out) {
  __shared__ __align__(16) unsigned short tile[2][RT][TSTRIDE];

  const int tid   = threadIdx.x;
  const int lane  = tid & 31;
  const int wave  = tid >> 5;                  // o-tile index 0..7
  const int mBase = blockIdx.x * (RT * 16);

  // ---- staging role: thread -> (row srow of each of the RT tiles, 16B run)
  const int srow = tid >> 4;                   // 0..15
  const int scol = (tid & 15) * 8;             // element column
  const unsigned short* xb[RT];
  const int*            np[RT];
  bool                  rok[RT];
#pragma unroll
  for (int tt = 0; tt < RT; ++tt) {
    const int m = mBase + tt * 16 + srow;
    rok[tt] = (m < MTOT);
    const int mc = rok[tt] ? m : 0;
    const int sb = mc / NN;
    const int sn = mc - sb * NN;
    xb[tt] = xT + (size_t)sb * NN * CIN;
    np[tt] = nbr + (size_t)sn * KNB;
  }

  // ---- compute role
  const int o  = wave * 16 + (lane & 15);
  const float bo = bias[o];
  const int arow = lane & 15;
  const int aoff = (lane < 16) ? 0 : 8;        // A-frag 16-bit layout

  // prologue: gather k = 0 and store to buffer 0
  uint4 st[RT];
#pragma unroll
  for (int tt = 0; tt < RT; ++tt) {
    const int nb = rok[tt] ? np[tt][0] : -1;
    st[tt] = make_uint4(0u, 0u, 0u, 0u);
    if (nb >= 0) st[tt] = *(const uint4*)(xb[tt] + (size_t)nb * CIN + scol);
  }
#pragma unroll
  for (int tt = 0; tt < RT; ++tt)
    *(uint4*)&tile[0][tt][srow * APITCH + scol] = st[tt];

  v8f acc[RT];
#pragma unroll
  for (int tt = 0; tt < RT; ++tt)
    acc[tt] = (v8f){0.f, 0.f, 0.f, 0.f, 0.f, 0.f, 0.f, 0.f};

  for (int k = 0; k < KNB; ++k) {
    __syncthreads();                           // tile[k&1] ready everywhere
    // issue next gather early: global latency overlaps the 16 WMMAs
    if (k + 1 < KNB) {
#pragma unroll
      for (int tt = 0; tt < RT; ++tt) {
        const int nb = rok[tt] ? np[tt][k + 1] : -1;
        st[tt] = make_uint4(0u, 0u, 0u, 0u);
        if (nb >= 0) st[tt] = *(const uint4*)(xb[tt] + (size_t)nb * CIN + scol);
      }
    }
    const unsigned short* paBase = &tile[k & 1][0][arow * APITCH + aoff];

    // all 4 B fragments for this k: one 8x global_load_b128 clause
    Frag ub[4];
#pragma unroll
    for (int ii = 0; ii < 4; ++ii) {
      const unsigned short* pb =
          wPack + ((size_t)(((k * 4 + ii) * 8 + wave) * 32 + lane)) * 16;
      ub[ii].q[0] = *(const uint4*)(pb);
      ub[ii].q[1] = *(const uint4*)(pb + 8);
    }

    // A-fragment software pipeline, distance 3 (3 rotating slots)
    Frag fa[3];
#pragma unroll
    for (int s = 0; s < 3; ++s) fa[s] = ldA(paBase, s);
#pragma unroll
    for (int s = 0; s < 16; ++s) {
      const int tt = s & 3;
      const int ii = s >> 2;
      acc[tt] = __builtin_amdgcn_wmma_f32_16x16x32_bf16(
          /*neg_a=*/false, fa[s % 3].v, /*neg_b=*/false, ub[ii].v,
          /*c_mod=*/(short)0, acc[tt], /*reuse_a=*/false, /*reuse_b=*/false);
      if (s + 3 < 16) fa[s % 3] = ldA(paBase, s + 3);
    }

    if (k + 1 < KNB) {  // safe: all waves passed this iteration's barrier,
#pragma unroll          // so nobody still reads buffer (k+1)&1
      for (int tt = 0; tt < RT; ++tt)
        *(uint4*)&tile[(k + 1) & 1][tt][srow * APITCH + scol] = st[tt];
    }
  }

  // ---- epilogue: D layout -> lane holds 8 consecutive-n f32 for one o.
  // (VGPR r: lanes 0-15 -> M=r, lanes 16-31 -> M=8+r; col = lane%16 = o.)
#pragma unroll
  for (int tt = 0; tt < RT; ++tt) {
    const int mt = mBase + tt * 16;
    if (mt >= MTOT) continue;
    const int b0 = mt / NN;
    if (b0 == (mt + 15) / NN) {                // tile doesn't straddle b
      const int n0 = (mt - b0 * NN) + ((lane < 16) ? 0 : 8);
      float* po = out + ((size_t)b0 * COUT + o) * NN + n0;
#pragma unroll
      for (int r = 0; r < 8; ++r) po[r] = acc[tt][r] + bo;
    } else {
      const int mr = mt + ((lane < 16) ? 0 : 8);
#pragma unroll
      for (int r = 0; r < 8; ++r) {
        const int mm = mr + r;
        const int bb = mm / NN;
        const int nn = mm - bb * NN;
        out[((size_t)bb * COUT + o) * NN + nn] = acc[tt][r] + bo;
      }
    }
  }
}

// --------------------------------------------------------------------------
extern "C" void kernel_launch(void* const* d_in, const int* in_sizes, int n_in,
                              void* d_out, int out_size, void* d_ws, size_t ws_size,
                              hipStream_t stream) {
  const float* x    = (const float*)d_in[0];   // (B, CIN, N) f32
  const float* w    = (const float*)d_in[1];   // (COUT, CIN, K) f32
  const float* bias = (const float*)d_in[2];   // (COUT,) f32
  const int*   nbr  = (const int*)d_in[3];     // (N, K) int
  float* out = (float*)d_out;                  // (B, COUT, N) f32

  unsigned short* xT = (unsigned short*)d_ws;                       // 15.2 MB
  unsigned short* wp = (unsigned short*)((char*)d_ws +
                        (size_t)BB * NN * CIN * sizeof(unsigned short)); // 623 KB

  dim3 tgrid((NN + 31) / 32, CIN / 32, BB);
  transpose_x_kernel<<<tgrid, dim3(32, 8), 0, stream>>>(x, xT);

  const int wtotal = KNB * 4 * 8 * 32 * 16;
  pack_w_kernel<<<(wtotal + 255) / 256, 256, 0, stream>>>(w, wp);

  const int nblk = (MTOT / 16 + RT - 1) / RT;  // 928 (last WG: 2 row-tiles)
  conv_main_kernel<<<nblk, 256, 0, stream>>>(xT, wp, nbr, bias, out);
}
